// SwinTransformerBlockFlex_3788161155764
// MI455X (gfx1250) — compile-verified, matrix operand order checked by reading, not run on hardware
//
#include <hip/hip_runtime.h>
#include <hip/hip_bf16.h>
#include <math.h>

// ---------------------------------------------------------------------------
// Swin-V2 block (window-partitioned full attention, cosine attn + RoPE + MLP)
// for MI455X / gfx1250: all GEMMs + both attention GEMMs via
// v_wmma_f32_16x16x32_f16 (wave32), f32 softmax/LN statistics.
// Tile movement: GLOBAL_LOAD_ASYNC_TO_LDS_B128 (ASYNCcnt) when the toolchain
// exposes it, double-buffered so DMA overlaps WMMA; VGPR-path fallback kept.
// ---------------------------------------------------------------------------

typedef _Float16 h8  __attribute__((ext_vector_type(8)));
typedef _Float16 h16 __attribute__((ext_vector_type(16)));
typedef float    f8  __attribute__((ext_vector_type(8)));
typedef int      v4i_t __attribute__((vector_size(16)));

union Frag16 { h16 v; h8 h[2]; _Float16 e[16]; };

#define NB    2
#define NH    12
#define CDIM  384
#define HD    32
#define SEQ   1764
#define SPAD  1792
#define MROWS 3528           // NB * SEQ
#define NTOK  49
#define HWDIM 42
#define LN100 4.6051701860f
#define INV_SQRT_HD 0.1767766952966369f

#if defined(__HIP_DEVICE_COMPILE__) && __has_builtin(__builtin_amdgcn_global_load_async_to_lds_b128)
#define USE_ASYNC_CP 1
#else
#define USE_ASYNC_CP 0
#endif

static __device__ inline f8 f8zero() {
  f8 z;
  for (int i = 0; i < 8; ++i) z[i] = 0.f;
  return z;
}

// 16-byte tile copy global -> LDS. Async (ASYNCcnt) on CDNA5 toolchains that
// expose the builtin; otherwise classic load + ds_store.
static __device__ inline void cp16(_Float16* l, const _Float16* g) {
#if USE_ASYNC_CP
  v4i_t* gg = (v4i_t*)g;   // generic bitcast (drops const)
  v4i_t* ll = (v4i_t*)l;
  __builtin_amdgcn_global_load_async_to_lds_b128(
      (__attribute__((address_space(1))) v4i_t*)gg,
      (__attribute__((address_space(3))) v4i_t*)ll, 0, 0);
#else
  *(h8*)l = *(const h8*)g;
#endif
}

static __device__ inline void cp_wait_all() {
#if USE_ASYNC_CP
#if __has_builtin(__builtin_amdgcn_s_wait_asynccnt)
  __builtin_amdgcn_s_wait_asynccnt(0);
#else
  asm volatile("s_wait_asynccnt 0x0" ::: "memory");
#endif
#endif
}

// Load one 16-half WMMA fragment for this lane following the CDNA5 16-bit
// A/B VGPR layout: lane<16 -> K {0..7,16..23}, lane>=16 -> K {8..15,24..31}.
// rowbase must be 16B aligned; offsets are 16B multiples.
static __device__ inline h16 ld_frag_row(const _Float16* rowbase, int lane) {
  Frag16 f;
  const int off = (lane & 16) ? 8 : 0;
  f.h[0] = *(const h8*)(rowbase + off);
  f.h[1] = *(const h8*)(rowbase + off + 16);
  return f.v;
}

// ---------------------------------------------------------------------------
// Elementwise helpers
// ---------------------------------------------------------------------------
__global__ __launch_bounds__(256) void k_f32_to_f16(const float* __restrict__ in,
                                                    _Float16* __restrict__ out, int n) {
  for (int i = blockIdx.x * 256 + threadIdx.x; i < n; i += gridDim.x * 256)
    out[i] = (_Float16)in[i];
}

__global__ __launch_bounds__(256) void k_zero_f16(_Float16* __restrict__ out, int n) {
  for (int i = blockIdx.x * 256 + threadIdx.x; i < n; i += gridDim.x * 256)
    out[i] = (_Float16)0.f;
}

// Window partition: x[B, L, C] f32 -> xw[B*Nw*N, C] f16
__global__ __launch_bounds__(256) void k_win_part(const float* __restrict__ x,
                                                  _Float16* __restrict__ xw) {
  const int total = MROWS * CDIM;
  for (int idx = blockIdx.x * 256 + threadIdx.x; idx < total; idx += gridDim.x * 256) {
    const int c = idx % CDIM;
    const int m = idx / CDIM;
    const int b = m / SEQ;
    const int s = m % SEQ;
    const int w = s / NTOK, n = s % NTOK;
    const int wr = w / 6, wc = w % 6;
    const int ii = n / 7, jj = n % 7;
    const int l = (wr * 7 + ii) * HWDIM + (wc * 7 + jj);
    xw[idx] = (_Float16)x[((size_t)b * SEQ + l) * CDIM + c];
  }
}

// ---------------------------------------------------------------------------
// Tiled WMMA GEMM: out[M,N] = A[M,K](f16) @ Bw[N,K]^T(f16) (+bias)(+GELU)
// Block = 256 thr (8 waves), 64x64 tile, each wave one 16x32 slice.
// Double-buffered LDS tiles; async copy for step i+1 overlaps WMMA of step i.
// ---------------------------------------------------------------------------
#define TILE_H 2560  // 64 rows * 40-half padded stride (conflict-free, 16B aligned)

__global__ __launch_bounds__(256) void k_gemm_wmma(
    const _Float16* __restrict__ A, const _Float16* __restrict__ Bw,
    const float* __restrict__ bias, float* __restrict__ outF,
    _Float16* __restrict__ outH, int M, int N, int K, int gelu) {
  __shared__ __align__(16) _Float16 As[2 * TILE_H];
  __shared__ __align__(16) _Float16 Bs[2 * TILE_H];
  const int tid  = threadIdx.x;
  const int lane = tid & 31;
  const int wave = tid >> 5;
  const int wm = wave & 3;          // 4 waves along M
  const int wn = wave >> 2;         // 2 waves along N
  const int m0 = blockIdx.y << 6;
  const int n0 = blockIdx.x << 6;
  const int lrow = tid >> 2;        // 0..63: cooperative tile row
  const int lseg = (tid & 3) << 3;  // 0,8,16,24 halves (16B chunks)
  const int loff = lrow * 40 + lseg;
  // Edge rows clamp to M-1: wrong values only reach output rows >= M,
  // which are never stored (WMMA output rows are independent).
  const int gm = (m0 + lrow < M) ? (m0 + lrow) : (M - 1);
  const _Float16* gA = A + (size_t)gm * K + lseg;
  const _Float16* gB = Bw + (size_t)(n0 + lrow) * K + lseg;
  const int nk = K >> 5;
  f8 acc0 = f8zero(), acc1 = f8zero();
  cp16(As + loff, gA);
  cp16(Bs + loff, gB);
  for (int i = 0; i < nk; ++i) {
    const int cur = (i & 1) * TILE_H;
    cp_wait_all();
    __syncthreads();
    if (i + 1 < nk) {
      const int nxt = ((i + 1) & 1) * TILE_H;
      cp16(As + nxt + loff, gA + (i + 1) * 32);
      cp16(Bs + nxt + loff, gB + (i + 1) * 32);
    }
    const h16 af = ld_frag_row(As + cur + (wm * 16 + (lane & 15)) * 40, lane);
    const h16 b0 = ld_frag_row(Bs + cur + (wn * 32 + (lane & 15)) * 40, lane);
    const h16 b1 = ld_frag_row(Bs + cur + (wn * 32 + 16 + (lane & 15)) * 40, lane);
    acc0 = __builtin_amdgcn_wmma_f32_16x16x32_f16(false, af, false, b0, (short)0, acc0, false, false);
    acc1 = __builtin_amdgcn_wmma_f32_16x16x32_f16(false, af, false, b1, (short)0, acc1, false, false);
  }
  // C/D layout: VGPR r, lane l -> row = r + 8*(l>=16), col = l%16
  const int col0  = n0 + wn * 32 + (lane & 15);
  const int rbase = m0 + wm * 16 + ((lane >> 4) << 3);
  for (int r = 0; r < 8; ++r) {
    const int gr = rbase + r;
    if (gr >= M) continue;
    float v0 = acc0[r], v1 = acc1[r];
    if (bias) { v0 += bias[col0]; v1 += bias[col0 + 16]; }
    if (gelu) {
      v0 = 0.5f * v0 * (1.f + erff(v0 * 0.7071067811865475f));
      v1 = 0.5f * v1 * (1.f + erff(v1 * 0.7071067811865475f));
    }
    if (outF) { outF[(size_t)gr * N + col0] = v0; outF[(size_t)gr * N + col0 + 16] = v1; }
    if (outH) { outH[(size_t)gr * N + col0] = (_Float16)v0; outH[(size_t)gr * N + col0 + 16] = (_Float16)v1; }
  }
}

// ---------------------------------------------------------------------------
// QKV split + bias + L2 norm (q,k) + logit scale + axial RoPE.
// One wave per (token, head); lane = channel d (0..31).
// ---------------------------------------------------------------------------
__global__ __launch_bounds__(128) void k_qkv_split_rope(
    const float* __restrict__ qkvraw, const float* __restrict__ q_bias,
    const float* __restrict__ v_bias, const float* __restrict__ logit_scale,
    _Float16* __restrict__ Qp, _Float16* __restrict__ Kp, _Float16* __restrict__ Vp) {
  const int gid  = blockIdx.x * 4 + (threadIdx.x >> 5);   // (token,head) pair
  const int lane = threadIdx.x & 31;                      // d
  if (gid >= MROWS * NH) return;                          // wave-uniform exit
  const int head = gid % NH;
  const int m    = gid / NH;
  const int b    = m / SEQ;
  const int s    = m % SEQ;
  const int n    = s % NTOK;
  const float* row = qkvraw + (size_t)m * (3 * CDIM);
  const int d = lane;
  float qv = row[head * HD + d] + q_bias[head * HD + d];
  float kv = row[CDIM + head * HD + d];
  float vv = row[2 * CDIM + head * HD + d] + v_bias[head * HD + d];
  // L2 norm over 32 lanes
  float qn = qv * qv, kn = kv * kv;
  for (int mk = 1; mk < 32; mk <<= 1) {
    qn += __shfl_xor(qn, mk, 32);
    kn += __shfl_xor(kn, mk, 32);
  }
  qv /= fmaxf(sqrtf(qn), 1e-12f);
  kv /= fmaxf(sqrtf(kn), 1e-12f);
  qv *= expf(fmaxf(logit_scale[head], LN100));
  // axial RoPE: d<16 -> row-axis freq, d>=16 -> col-axis; theta 10000, r=2
  const int pi = n / 7, pj = n % 7;
  const float pos = (d < 16) ? (float)pi : (float)pj;
  const float t = (float)((d & 15) >> 1);
  const float f = pos * expf(-t * 1.1512925464970229f);   // 10000^(-t/8)
  const float cs = cosf(f), sn = sinf(f);
  const float qp = __shfl_xor(qv, 1, 32);
  const float kp = __shfl_xor(kv, 1, 32);
  const float sgn = (d & 1) ? 1.f : -1.f;
  const float qr = qv * cs + sgn * qp * sn;
  const float kr = kv * cs + sgn * kp * sn;
  const size_t dst = ((size_t)(b * NH + head) * SPAD + s) * HD + d;
  Qp[dst] = (_Float16)qr;
  Kp[dst] = (_Float16)kr;
  Vp[dst] = (_Float16)vv;
}

// ---------------------------------------------------------------------------
// Attention: per (b,h), flash-style over S=1764, fixed-max softmax
// (scores bounded by scale/sqrt(hd) since q,k are unit + scaled).
// Block = 4 waves x 16 query rows; K/V chunks of 32 double-buffered in LDS.
// One barrier per chunk: P staging LDS is wave-private and same-wave LDS ops
// execute in order on CDNA5, so no extra barriers are needed for it.
// ---------------------------------------------------------------------------
#define KVT 1280  // 32 rows * 40-half padded stride

__global__ __launch_bounds__(128) void k_attention(
    const _Float16* __restrict__ Qp, const _Float16* __restrict__ Kp,
    const _Float16* __restrict__ Vp, const float* __restrict__ logit_scale,
    _Float16* __restrict__ Oh) {
  __shared__ __align__(16) _Float16 Ks[2 * KVT];
  __shared__ __align__(16) _Float16 Vs[2 * KVT];
  __shared__ __align__(16) _Float16 Ps[4 * 16 * 40];
  const int bh   = blockIdx.y;
  const int head = bh % NH;
  const int m0   = blockIdx.x << 6;
  const int tid  = threadIdx.x;
  const int lane = tid & 31;
  const int wave = tid >> 5;
  const _Float16* q = Qp + (size_t)bh * SPAD * HD;
  const _Float16* k = Kp + (size_t)bh * SPAD * HD;
  const _Float16* v = Vp + (size_t)bh * SPAD * HD;
  const float smax = expf(fmaxf(logit_scale[head], LN100)) * INV_SQRT_HD;  // exact bound
  const int qrow = m0 + wave * 16 + (lane & 15);
  const h16 qf = ld_frag_row(q + (size_t)qrow * HD, lane);
  f8 o0 = f8zero(), o1 = f8zero(), lacc = f8zero();
  _Float16* myP = Ps + wave * (16 * 40);
  const int lrow = tid >> 2;         // 0..31
  const int lseg = (tid & 3) << 3;   // 0,8,16,24
  const int loff = lrow * 40 + lseg;
  const _Float16* gK = k + (size_t)lrow * HD + lseg;
  const _Float16* gV = v + (size_t)lrow * HD + lseg;
  const int nch = SPAD / 32;         // 56
  cp16(Ks + loff, gK);
  cp16(Vs + loff, gV);
  for (int jt = 0; jt < nch; ++jt) {
    const int cur = (jt & 1) * KVT;
    cp_wait_all();
    __syncthreads();
    if (jt + 1 < nch) {
      const int nxt = ((jt + 1) & 1) * KVT;
      cp16(Ks + nxt + loff, gK + (size_t)(jt + 1) * 32 * HD);
      cp16(Vs + nxt + loff, gV + (size_t)(jt + 1) * 32 * HD);
    }
    // scores: S = Q(16x32) x K^T(32x16) for two 16-col subtiles
    const h16 kb0 = ld_frag_row(Ks + cur + (lane & 15) * 40, lane);
    const h16 kb1 = ld_frag_row(Ks + cur + (16 + (lane & 15)) * 40, lane);
    f8 s0 = __builtin_amdgcn_wmma_f32_16x16x32_f16(false, qf, false, kb0, (short)0, f8zero(), false, false);
    f8 s1 = __builtin_amdgcn_wmma_f32_16x16x32_f16(false, qf, false, kb1, (short)0, f8zero(), false, false);
    const int j0  = jt * 32;
    const int jc0 = j0 + (lane & 15);
    const int jc1 = jc0 + 16;
    const int prb = (lane >> 4) << 3;
    for (int r = 0; r < 8; ++r) {
      const float p0 = (jc0 < SEQ) ? expf(s0[r] * INV_SQRT_HD - smax) : 0.f;
      const float p1 = (jc1 < SEQ) ? expf(s1[r] * INV_SQRT_HD - smax) : 0.f;
      lacc[r] += p0 + p1;
      myP[(prb + r) * 40 + (lane & 15)]      = (_Float16)p0;
      myP[(prb + r) * 40 + 16 + (lane & 15)] = (_Float16)p1;
    }
    // O += P(16x32) x V(32x32): wave-private LDS transpose of P, then WMMA
    const h16 pf = ld_frag_row(myP + (lane & 15) * 40, lane);
    Frag16 v0f, v1f;
    const int jo = (lane & 16) ? 8 : 0;
    const int dc = lane & 15;
    for (int t = 0; t < 8; ++t) {
      v0f.e[t]     = Vs[cur + (jo + t) * 40 + dc];
      v0f.e[8 + t] = Vs[cur + (jo + 16 + t) * 40 + dc];
      v1f.e[t]     = Vs[cur + (jo + t) * 40 + 16 + dc];
      v1f.e[8 + t] = Vs[cur + (jo + 16 + t) * 40 + 16 + dc];
    }
    o0 = __builtin_amdgcn_wmma_f32_16x16x32_f16(false, pf, false, v0f.v, (short)0, o0, false, false);
    o1 = __builtin_amdgcn_wmma_f32_16x16x32_f16(false, pf, false, v1f.v, (short)0, o1, false, false);
  }
  // row-sum of exp across the 16-lane group holding each row
  for (int r = 0; r < 8; ++r) {
    float s = lacc[r];
    s += __shfl_xor(s, 1, 32);
    s += __shfl_xor(s, 2, 32);
    s += __shfl_xor(s, 4, 32);
    s += __shfl_xor(s, 8, 32);
    lacc[r] = 1.0f / fmaxf(s, 1e-30f);
  }
  const int orb = m0 + wave * 16 + ((lane >> 4) << 3);
  const int dc = lane & 15;
  for (int r = 0; r < 8; ++r) {
    const int gr = orb + r;
    if (gr < SEQ) {
      _Float16* op = Oh + ((size_t)bh * SEQ + gr) * HD;
      op[dc]      = (_Float16)(o0[r] * lacc[r]);
      op[16 + dc] = (_Float16)(o1[r] * lacc[r]);
    }
  }
}

// ---------------------------------------------------------------------------
// out = base + LayerNorm(ynew)  (optionally window-reverse gather of ynew)
// Block = 384 threads = one row of C channels.
// ---------------------------------------------------------------------------
__global__ __launch_bounds__(384) void k_residual_ln(
    const float* __restrict__ base, const float* __restrict__ ynew,
    const float* __restrict__ g, const float* __restrict__ bb,
    float* __restrict__ outF, _Float16* __restrict__ outH, int gather) {
  __shared__ float sred[12];
  __shared__ float stat[2];
  const int rowid = blockIdx.x;          // b*SEQ + l
  const int c = threadIdx.x;
  const int lane = c & 31, warp = c >> 5;
  size_t src;
  if (gather) {
    const int b = rowid / SEQ, l = rowid % SEQ;
    const int r = l / HWDIM, cc = l % HWDIM;
    const int w = (r / 7) * 6 + cc / 7;
    const int n = (r % 7) * 7 + cc % 7;
    src = (((size_t)b * 36 + w) * NTOK + n) * CDIM + c;
  } else {
    src = (size_t)rowid * CDIM + c;
  }
  const float vv = ynew[src];
  float s = vv;
  for (int mk = 1; mk < 32; mk <<= 1) s += __shfl_xor(s, mk, 32);
  if (lane == 0) sred[warp] = s;
  __syncthreads();
  if (c == 0) {
    float t = 0.f;
    for (int i = 0; i < 12; ++i) t += sred[i];
    stat[0] = t * (1.f / CDIM);
  }
  __syncthreads();
  const float mu = stat[0];
  float dv = (vv - mu) * (vv - mu);
  for (int mk = 1; mk < 32; mk <<= 1) dv += __shfl_xor(dv, mk, 32);
  if (lane == 0) sred[warp] = dv;
  __syncthreads();
  if (c == 0) {
    float t = 0.f;
    for (int i = 0; i < 12; ++i) t += sred[i];
    stat[1] = t * (1.f / CDIM);
  }
  __syncthreads();
  const float y = (vv - mu) * rsqrtf(stat[1] + 1e-5f) * g[c] + bb[c];
  const float o = base[(size_t)rowid * CDIM + c] + y;
  outF[(size_t)rowid * CDIM + c] = o;
  if (outH) outH[(size_t)rowid * CDIM + c] = (_Float16)o;
}

// ---------------------------------------------------------------------------
// Host orchestration
// ---------------------------------------------------------------------------
static inline size_t al256(size_t x) { return (x + 255) & ~(size_t)255; }

extern "C" void kernel_launch(void* const* d_in, const int* in_sizes, int n_in,
                              void* d_out, int out_size, void* d_ws, size_t ws_size,
                              hipStream_t stream) {
  const float* x           = (const float*)d_in[0];
  const float* qkv_w       = (const float*)d_in[1];
  const float* q_bias      = (const float*)d_in[2];
  const float* v_bias      = (const float*)d_in[3];
  const float* logit_scale = (const float*)d_in[4];
  const float* proj_w      = (const float*)d_in[5];
  const float* proj_b      = (const float*)d_in[6];
  const float* norm1_g     = (const float*)d_in[7];
  const float* norm1_b     = (const float*)d_in[8];
  const float* fc1_w       = (const float*)d_in[9];
  const float* fc1_b       = (const float*)d_in[10];
  const float* fc2_w       = (const float*)d_in[11];
  const float* fc2_b       = (const float*)d_in[12];
  const float* norm2_g     = (const float*)d_in[13];
  const float* norm2_b     = (const float*)d_in[14];
  float* out = (float*)d_out;

  // ---- workspace bump allocator (~63 MB total) ----
  char* w = (char*)d_ws;
  auto alloc = [&](size_t bytes) { char* p = w; w += al256(bytes); return p; };
  _Float16* Wqkv = (_Float16*)alloc((size_t)3 * CDIM * CDIM * 2);
  _Float16* Wprj = (_Float16*)alloc((size_t)CDIM * CDIM * 2);
  _Float16* Wfc1 = (_Float16*)alloc((size_t)4 * CDIM * CDIM * 2);
  _Float16* Wfc2 = (_Float16*)alloc((size_t)4 * CDIM * CDIM * 2);
  _Float16* XW   = (_Float16*)alloc((size_t)MROWS * CDIM * 2);
  float*    QKVr = (float*)   alloc((size_t)MROWS * 3 * CDIM * 4);
  const size_t padElems = (size_t)NB * NH * SPAD * HD;
  _Float16* Qp   = (_Float16*)alloc(3 * padElems * 2);
  _Float16* Kp   = Qp + padElems;
  _Float16* Vp   = Kp + padElems;
  _Float16* AttO = (_Float16*)alloc((size_t)MROWS * CDIM * 2);
  float*    Pout = (float*)   alloc((size_t)MROWS * CDIM * 4);
  float*    X1f  = (float*)   alloc((size_t)MROWS * CDIM * 4);
  _Float16* X1h  = (_Float16*)alloc((size_t)MROWS * CDIM * 2);
  _Float16* Y1h  = (_Float16*)alloc((size_t)MROWS * 4 * CDIM * 2);
  float*    Y2f  = (float*)   alloc((size_t)MROWS * CDIM * 4);
  (void)ws_size; (void)n_in; (void)in_sizes; (void)out_size;

  // 1) weights -> f16
  k_f32_to_f16<<<(3 * CDIM * CDIM + 255) / 256, 256, 0, stream>>>(qkv_w, Wqkv, 3 * CDIM * CDIM);
  k_f32_to_f16<<<(CDIM * CDIM + 255) / 256, 256, 0, stream>>>(proj_w, Wprj, CDIM * CDIM);
  k_f32_to_f16<<<(4 * CDIM * CDIM + 255) / 256, 256, 0, stream>>>(fc1_w, Wfc1, 4 * CDIM * CDIM);
  k_f32_to_f16<<<(4 * CDIM * CDIM + 255) / 256, 256, 0, stream>>>(fc2_w, Wfc2, 4 * CDIM * CDIM);

  // 2) window partition -> f16 A
  k_win_part<<<(MROWS * CDIM + 255) / 256, 256, 0, stream>>>(x, XW);

  // 3) QKV GEMM  [3528,384] x [1152,384]^T -> f32
  k_gemm_wmma<<<dim3(3 * CDIM / 64, (MROWS + 63) / 64), 256, 0, stream>>>(
      XW, Wqkv, nullptr, QKVr, nullptr, MROWS, 3 * CDIM, CDIM, 0);

  // 4) zero padded q/k/v then split + normalize + scale + RoPE
  k_zero_f16<<<(int)((3 * padElems + 255) / 256), 256, 0, stream>>>(Qp, (int)(3 * padElems));
  k_qkv_split_rope<<<(MROWS * NH + 3) / 4, 128, 0, stream>>>(
      QKVr, q_bias, v_bias, logit_scale, Qp, Kp, Vp);

  // 5) attention (24 bh, 28 m-tiles of 64 rows)
  k_attention<<<dim3((SEQ + 63) / 64, NB * NH), 128, 0, stream>>>(Qp, Kp, Vp, logit_scale, AttO);

  // 6) proj GEMM (+bias) -> f32
  k_gemm_wmma<<<dim3(CDIM / 64, (MROWS + 63) / 64), 256, 0, stream>>>(
      AttO, Wprj, proj_b, Pout, nullptr, MROWS, CDIM, CDIM, 0);

  // 7) x1 = x + LN(window_reverse(proj_out))
  k_residual_ln<<<MROWS, 384, 0, stream>>>(x, Pout, norm1_g, norm1_b, X1f, X1h, 1);

  // 8) fc1 GEMM + bias + GELU -> f16
  k_gemm_wmma<<<dim3(4 * CDIM / 64, (MROWS + 63) / 64), 256, 0, stream>>>(
      X1h, Wfc1, fc1_b, nullptr, Y1h, MROWS, 4 * CDIM, CDIM, 1);

  // 9) fc2 GEMM + bias -> f32
  k_gemm_wmma<<<dim3(CDIM / 64, (MROWS + 63) / 64), 256, 0, stream>>>(
      Y1h, Wfc2, fc2_b, Y2f, nullptr, MROWS, CDIM, 4 * CDIM, 0);

  // 10) out = x1 + LN(y2)
  k_residual_ln<<<MROWS, 384, 0, stream>>>(X1f, Y2f, norm2_g, norm2_b, out, nullptr, 0);
}
